// Sequence_11931419149125
// MI455X (gfx1250) — matrix operand order repaired and checked by reading, not compiled, not run on hardware
//
#include <hip/hip_runtime.h>

// ---------------- problem constants ----------------
#define Bsz   4096
#define Tlen  2048
#define Hd    51
#define G4    204     // 4*H
#define NT    13      // n-tiles of 16 covering 208
#define NP    208     // padded gate width (4 gates x 52)
#define GS    52      // per-gate stride in remapped gate layout
#define CH    64      // t-chunk for x staging / out flushing

typedef __attribute__((ext_vector_type(16))) __bf16 v16bf;
typedef __attribute__((ext_vector_type(8)))  float  v8f;

union Frag {
    v16bf v;
    uint4 q[2];
    unsigned int u[8];
};

// hardware f32 -> bf16 (RNE) via native __bf16 converts
__device__ __forceinline__ unsigned int pack_bf16(float a, float b) {
    union { __bf16 h[2]; unsigned int u; } p;
    p.h[0] = (__bf16)a;
    p.h[1] = (__bf16)b;
    return p.u;
}
__device__ __forceinline__ unsigned short bf16_1(float a) {
    union { __bf16 h; unsigned short u; } p;
    p.h = (__bf16)a;
    return p.u;
}

__device__ __forceinline__ float tanh1(float x) {
#if __has_builtin(__builtin_amdgcn_tanhf)
    return __builtin_amdgcn_tanhf(x);       // gfx1250 v_tanh_f32
#else
    x = fminf(fmaxf(x, -10.0f), 10.0f);
    float e = __expf(2.0f * x);
    return (e - 1.0f) / (e + 1.0f);
#endif
}
__device__ __forceinline__ float sigm(float x) {
    return 0.5f * tanh1(0.5f * x) + 0.5f;
}

__global__ __launch_bounds__(32)
void lstm_wave_kernel(const float* __restrict__ x,
                      const float* __restrict__ Wih,
                      const float* __restrict__ Whh,
                      const float* __restrict__ bih,
                      const float* __restrict__ bhh,
                      const float* __restrict__ Wlin,
                      const float* __restrict__ blin,
                      float* __restrict__ out)
{
    // ---- LDS (all wave-private; one wave per block => no barriers) ----
    __shared__ unsigned int sH[16 * 32];     // [h, x] state, bf16 pairs, K padded to 64 (k=51 holds x_t)
    __shared__ float        sG[16 * NP];     // gates scratch (f32), remapped: gate g at [g*52, g*52+52)
    __shared__ float        sX[16 * CH];     // x chunk
    __shared__ float        sY[2 * 16 * CH]; // output chunk (ch0: x-y, ch1: y)
    __shared__ float        sWlin[52];       // W_lin padded

    const int l  = threadIdx.x;    // 0..31
    const int lm = l & 15;
    const int hi = l >> 4;
    const int b0 = blockIdx.x * 16;

    // combined weight element for remapped column np (0..207) and k (0..63):
    //   gate g = np/52, col c = np%52; orig row r = g*51+c (c<51), else zero pad
    //   k<51 -> Whh[r][k];  k==51 -> Wih[r]  (x_t folded in at k=51);  else 0
    auto wval = [&](int np, int k) -> float {
        int g = np / GS;
        int c = np - g * GS;
        if (c >= Hd) return 0.0f;
        int r = g * Hd + c;
        if (k < Hd)  return Whh[r * Hd + k];
        if (k == Hd) return Wih[r];
        return 0.0f;
    };

    // ---------- B fragments resident in VGPRs for the whole kernel (208 VGPRs) ----------
    // fragment (tile j, slice s): this lane holds n = 16j+lm,
    // dword v = bf16 pair (k = 32s + 16(v/4) + 8hi + 2(v%4), +1)
    Frag bw[NT][2];
#pragma unroll
    for (int j = 0; j < NT; ++j)
#pragma unroll
        for (int s = 0; s < 2; ++s)
#pragma unroll
            for (int v = 0; v < 8; ++v) {
                int np = j * 16 + lm;
                int k  = s * 32 + (v >> 2) * 16 + hi * 8 + (v & 3) * 2;
                bw[j][s].u[v] = pack_bf16(wval(np, k), wval(np, k + 1));
            }

    // bias folded into WMMA C operand (remapped columns)
    float biasv[NT];
#pragma unroll
    for (int j = 0; j < NT; ++j) {
        int np = j * 16 + lm;
        int g = np / GS, c = np - g * GS;
        biasv[j] = (c < Hd) ? (bih[g * Hd + c] + bhh[g * Hd + c]) : 0.0f;
    }
    const float bl = blin[0];

    for (int i = l; i < 52; i += 32) sWlin[i] = (i < Hd) ? Wlin[i] : 0.0f;
    for (int i = l; i < 16 * 32; i += 32) sH[i] = 0u;   // h0 = 0 (incl. padding)

    // cell state: hidden col = it*4 + hi*2 + e, e in {0,1}
    float cst[26];
#pragma unroll
    for (int i = 0; i < 26; ++i) cst[i] = 0.0f;

    for (int t0 = 0; t0 < Tlen; t0 += CH) {
        // ---- stage x[b0..b0+15, t0..t0+63]: lane (lm,hi) streams half a row ----
        {
            const float* xp = x + (size_t)(b0 + lm) * Tlen + t0 + hi * 32;
            float* dst = &sX[lm * CH + hi * 32];
#pragma unroll
            for (int c = 0; c < 32; c += 4)
                *(float4*)(dst + c) = *(const float4*)(xp + c);
        }

#pragma unroll 1
        for (int tc = 0; tc < CH; ++tc) {
            const float xv = sX[lm * CH + tc];

            // inject x_t as bf16 at k=51 (halfword 51 of this row); lane pairs
            // (l, l+16) store the same value to the same address (benign)
            ((unsigned short*)sH)[lm * 64 + 51] = bf16_1(xv);

            // ---------------- GEMM: gates = [h,x] @ [Whh;Wih]^T + bias ----------------
            Frag a0, a1;
            {
                const uint4* hp = (const uint4*)sH;   // 8 uint4 per row
                a0.q[0] = hp[lm * 8 + hi];
                a0.q[1] = hp[lm * 8 + hi + 2];
                a1.q[0] = hp[lm * 8 + 4 + hi];
                a1.q[1] = hp[lm * 8 + 4 + hi + 2];
            }
#pragma unroll
            for (int j = 0; j < NT; ++j) {
                v8f acc;
#pragma unroll
                for (int r = 0; r < 8; ++r) acc[r] = biasv[j];
                acc = __builtin_amdgcn_wmma_f32_16x16x32_bf16(false, a0.v, false, bw[j][0].v,
                                                              (short)0, acc, false, false);
                acc = __builtin_amdgcn_wmma_f32_16x16x32_bf16(false, a1.v, false, bw[j][1].v,
                                                              (short)0, acc, false, false);
                // D layout: VGPR r -> row r + 8*hi, lane low nibble -> col
#pragma unroll
                for (int r = 0; r < 8; ++r)
                    sG[(r + hi * 8) * NP + j * 16 + lm] = acc[r];
            }

            // ---------------- elementwise LSTM cell ----------------
            float yacc = 0.0f;
            const float* grow = &sG[lm * NP];
#pragma unroll
            for (int it = 0; it < 13; ++it) {
                const int hb = it * 4 + hi * 2;        // hidden cols hb, hb+1 (hb even)
                float2 gi = *(const float2*)(grow + hb);            // i-gate (8B aligned)
                float2 gf = *(const float2*)(grow + GS + hb);       // f-gate
                float2 gg = *(const float2*)(grow + 2 * GS + hb);   // g-gate
                float2 go = *(const float2*)(grow + 3 * GS + hb);   // o-gate

                float c0 = cst[it * 2 + 0], c1 = cst[it * 2 + 1];
                c0 = sigm(gf.x) * c0 + sigm(gi.x) * tanh1(gg.x);
                c1 = sigm(gf.y) * c1 + sigm(gi.y) * tanh1(gg.y);
                float h0 = sigm(go.x) * tanh1(c0);
                float h1 = sigm(go.y) * tanh1(c1);
                if (hb + 1 >= Hd) { h1 = 0.0f; c1 = 0.0f; }  // keep K-pad column zero
                cst[it * 2 + 0] = c0; cst[it * 2 + 1] = c1;

                sH[lm * 32 + (hb >> 1)] = pack_bf16(h0, h1); // h_new back to bf16 state
                float2 wl = *(const float2*)(&sWlin[hb]);
                yacc += h0 * wl.x + h1 * wl.y;
            }
            yacc += __shfl_xor(yacc, 16, 32);   // combine column-parity halves
            const float y = yacc + bl;
            // lane writes its channel (hi); pair lanes agree on xv and y
            sY[hi * 16 * CH + lm * CH + tc] = hi ? y : (xv - y);
        }

        // ---- flush: lane l owns (channel hi, row lm): 64 contiguous floats ----
        {
            float* op = out + (size_t)(b0 + lm) * 2 * Tlen + hi * Tlen + t0;
            const float* sp = &sY[hi * 16 * CH + lm * CH];
#pragma unroll
            for (int c = 0; c < CH; c += 4)
                *(float4*)(op + c) = *(const float4*)(sp + c);
        }
    }
}

extern "C" void kernel_launch(void* const* d_in, const int* in_sizes, int n_in,
                              void* d_out, int out_size, void* d_ws, size_t ws_size,
                              hipStream_t stream) {
    (void)in_sizes; (void)n_in; (void)out_size; (void)d_ws; (void)ws_size;
    const float* x    = (const float*)d_in[0];
    const float* Wih  = (const float*)d_in[1];
    const float* Whh  = (const float*)d_in[2];
    const float* bih  = (const float*)d_in[3];
    const float* bhh  = (const float*)d_in[4];
    const float* Wlin = (const float*)d_in[5];
    const float* blin = (const float*)d_in[6];
    float* out = (float*)d_out;

    dim3 grid(Bsz / 16);   // 256 independent 16-row recurrences, one wave each
    dim3 block(32);        // exactly one wave32 => no intra-step barriers needed
    lstm_wave_kernel<<<grid, block, 0, stream>>>(x, Wih, Whh, bih, bhh, Wlin, blin, out);
}
